// BaselineGPT2_65206193488068
// MI455X (gfx1250) — matrix operand order repaired and checked by reading, not compiled, not run on hardware
//
#include <hip/hip_runtime.h>
#include <hip/hip_bf16.h>
#include <math.h>

typedef __attribute__((ext_vector_type(16))) _Float16 v16h;
typedef __attribute__((ext_vector_type(8)))  _Float16 v8h;
typedef __attribute__((ext_vector_type(8)))  float    v8f;
typedef __attribute__((ext_vector_type(4)))  float    v4f;

#define SD   (2048 * 768)
#define SEQ  2048
#define DM   768
#define FFD  3072
#define NH   12
#define HDIM 64
#define VOC  50304

#if __has_builtin(__builtin_amdgcn_global_load_async_to_lds_b32) && \
    __has_builtin(__builtin_amdgcn_s_wait_asynccnt)
#define ASYNC_LDS 1
typedef __attribute__((address_space(1))) int* gptr_t;
typedef __attribute__((address_space(3))) int* lptr_t;
#else
#define ASYNC_LDS 0
#endif

// ---- WMMA fragment index maps (CDNA5 ISA 7.12.2, wave32) ----
// 16-bit A 16x32: lane row M=lane&15; elems 0..7 -> K={0..7}+8*half, elems 8..15 -> K={16..23}+8*half
//   => per lane: TWO contiguous 8-elem runs -> b128 loads
// 16-bit B 32x16: lane col N=lane&15; same K map as A (k_gemm uses K=e+16*half for A and B)
//   => per lane: ONE contiguous 16-elem run if tile is stored N-major (transposed)
// 32-bit C/D 16x16: lane -> N=lane&15; vgpr r -> M = r + 8*half

__device__ __forceinline__ v16h concat8(v8h lo, v8h hi) {
  return __builtin_shufflevector(lo, hi, 0,1,2,3,4,5,6,7,8,9,10,11,12,13,14,15);
}
__device__ __forceinline__ v8h cvt8(v4f a, v4f b) {
  v8h r;
  r[0]=(_Float16)a[0]; r[1]=(_Float16)a[1]; r[2]=(_Float16)a[2]; r[3]=(_Float16)a[3];
  r[4]=(_Float16)b[0]; r[5]=(_Float16)b[1]; r[6]=(_Float16)b[2]; r[7]=(_Float16)b[3];
  return r;
}
__device__ __forceinline__ float gelu_exact(float x) {
  return 0.5f * x * (1.0f + erff(x * 0.70710678118654752f));
}

// =====================  embedding  =====================
__global__ __launch_bounds__(256) void k_embed(const int* __restrict__ ids,
                                               const float* __restrict__ wte,
                                               const float* __restrict__ wpe,
                                               float* __restrict__ x) {
  int s = blockIdx.x;
  int tok = ids[s];
  for (int i = threadIdx.x; i < DM; i += 256)
    x[(size_t)s * DM + i] = wte[(size_t)tok * DM + i] + wpe[(size_t)s * DM + i];
}

// =====================  layernorm  =====================
__global__ __launch_bounds__(256) void k_layernorm(const float* __restrict__ x,
                                                   const float* __restrict__ g,
                                                   const float* __restrict__ b,
                                                   float* __restrict__ y) {
  __shared__ float red[256];
  int row = blockIdx.x, tid = threadIdx.x;
  const float* xr = x + (size_t)row * DM;
  float s = 0.f;
  for (int i = tid; i < DM; i += 256) s += xr[i];
  red[tid] = s; __syncthreads();
  for (int off = 128; off > 0; off >>= 1) {
    if (tid < off) red[tid] += red[tid + off];
    __syncthreads();
  }
  float mean = red[0] * (1.0f / DM);
  __syncthreads();
  float v = 0.f;
  for (int i = tid; i < DM; i += 256) { float d = xr[i] - mean; v += d * d; }
  red[tid] = v; __syncthreads();
  for (int off = 128; off > 0; off >>= 1) {
    if (tid < off) red[tid] += red[tid + off];
    __syncthreads();
  }
  float rstd = rsqrtf(red[0] * (1.0f / DM) + 1e-5f);
  for (int i = tid; i < DM; i += 256)
    y[(size_t)row * DM + i] = (xr[i] - mean) * rstd * g[i] + b[i];
}

// =====================  WMMA GEMM  =====================
// C[M,N] = A[M,K] @ B[K,N] (+bias) (gelu?) (+res).  M,N multiples of 128; K multiple of 32.
// 256 threads = 8 waves; block tile 128x128; wave tile 32x64 -> 8 WMMAs per wave per K-step.
// Double-buffered: A f16 staged sync (b128+cvt), B fp32 staged TRANSPOSED via async-to-LDS DMA.
#define GELU_FLAG 1
#define BM 128
#define BN 128
#define BK 32
#define LDA 48   // A LDS row stride (halves): 96 B, 16B multiple
#define LDB 36   // B LDS row stride (floats): 144 B, 16B multiple

__device__ __forceinline__ void stage_a(const float* __restrict__ A, _Float16 (*As)[LDA],
                                        int bm, int K, int k0, int tid) {
#pragma unroll
  for (int j = 0; j < 2; ++j) {
    int ch = j * 256 + tid;            // 512 chunks of 8
    int r = ch >> 2, co = (ch & 3) * 8;
    const float* src = A + (size_t)(bm + r) * K + (k0 + co);
    v4f a0 = *(const v4f*)src;
    v4f a1 = *(const v4f*)(src + 4);
    *(v8h*)&As[r][co] = cvt8(a0, a1);
  }
}

__device__ __forceinline__ void stage_b(const float* __restrict__ B, float (*Bs)[LDB],
                                        int bn, int N, int k0, int tid) {
#pragma unroll
  for (int j = 0; j < 2; ++j) {
    int ch = j * 256 + tid;            // 512 chunks (one n, 8 K-values each)
    int kg = ch >> 7, n = ch & 127;
#if ASYNC_LDS
#pragma unroll
    for (int t8 = 0; t8 < 8; ++t8)     // per-lane DMA transpose: B[k][n] -> Bst[n][k]
      __builtin_amdgcn_global_load_async_to_lds_b32(
          (gptr_t)&B[(size_t)(k0 + kg * 8 + t8) * N + (bn + n)],
          (lptr_t)&Bs[n][kg * 8 + t8], 0, 0);
#else
#pragma unroll
    for (int t8 = 0; t8 < 8; ++t8)
      Bs[n][kg * 8 + t8] = B[(size_t)(k0 + kg * 8 + t8) * N + (bn + n)];
#endif
  }
}

__global__ __launch_bounds__(256) void k_gemm(const float* __restrict__ A,
                                              const float* __restrict__ B,
                                              const float* __restrict__ bias,
                                              const float* __restrict__ res,
                                              float* __restrict__ C,
                                              int M, int N, int K, int flags) {
  __shared__ __align__(16) _Float16 Asl[2][BM][LDA];  // f16 row-major, double buffered
  __shared__ __align__(16) float    Bsl[2][BN][LDB];  // fp32 TRANSPOSED [n][k], double buffered
  int tid = threadIdx.x;
  int lane = tid & 31, wave = tid >> 5;
  int lh = lane >> 4, lm = lane & 15;
  int bm = blockIdx.x * BM;
  int bn = blockIdx.y * BN;
  int wm = (wave & 3) * 32;
  int wn = (wave >> 2) * 64;

  v8f acc[2][4];
#pragma unroll
  for (int mi = 0; mi < 2; ++mi)
#pragma unroll
    for (int ni = 0; ni < 4; ++ni) acc[mi][ni] = v8f{};

  int nk = K / BK;
  stage_a(A, Asl[0], bm, K, 0, tid);
  stage_b(B, Bsl[0], bn, N, 0, tid);
#if ASYNC_LDS
  __builtin_amdgcn_s_wait_asynccnt(0);
#endif
  __syncthreads();

  for (int t = 0; t < nk; ++t) {
    int cur = t & 1;
    if (t + 1 < nk) {                          // stage next tile into alternate buffers
      stage_a(A, Asl[cur ^ 1], bm, K, (t + 1) * BK, tid);
      stage_b(B, Bsl[cur ^ 1], bn, N, (t + 1) * BK, tid);
    }
    if (t + 2 < nk)                            // gfx1250 global_prefetch_b8, 2 tiles ahead
      __builtin_prefetch(&A[(size_t)(bm + (tid >> 1)) * K + (t + 2) * BK + (tid & 1) * 16], 0, 0);

    // ---- fragments from current buffer: aligned ds_load_b128 only
    v16h af[2], bf[4];
#pragma unroll
    for (int mi = 0; mi < 2; ++mi) {
      int row = wm + mi * 16 + lm;
      af[mi] = concat8(*(const v8h*)&Asl[cur][row][16 * lh],
                       *(const v8h*)&Asl[cur][row][16 * lh + 8]);
    }
#pragma unroll
    for (int ni = 0; ni < 4; ++ni) {
      int col = wn + ni * 16 + lm;
      const float* bp = &Bsl[cur][col][16 * lh];     // contiguous K run (K=e+16*half, same order as A)
      bf[ni] = concat8(cvt8(*(const v4f*)(bp),     *(const v4f*)(bp + 4)),
                       cvt8(*(const v4f*)(bp + 8), *(const v4f*)(bp + 12)));
    }
#pragma unroll
    for (int mi = 0; mi < 2; ++mi)
#pragma unroll
      for (int ni = 0; ni < 4; ++ni)
        acc[mi][ni] = __builtin_amdgcn_wmma_f32_16x16x32_f16(
            false, af[mi], false, bf[ni], (short)0, acc[mi][ni], false, false);

#if ASYNC_LDS
    __builtin_amdgcn_s_wait_asynccnt(0);       // next-tile DMA complete (this wave)
#endif
    __syncthreads();                           // all waves' staging complete
  }

  // ---- epilogue
#pragma unroll
  for (int mi = 0; mi < 2; ++mi)
#pragma unroll
    for (int ni = 0; ni < 4; ++ni) {
      int gn = bn + wn + ni * 16 + lm;
      float bb = bias ? bias[gn] : 0.f;
#pragma unroll
      for (int r = 0; r < 8; ++r) {
        int gm = bm + wm + mi * 16 + r + 8 * lh;
        float v = acc[mi][ni][r] + bb;
        if (flags & GELU_FLAG) v = gelu_exact(v);
        if (res) v += res[(size_t)gm * N + gn];
        C[(size_t)gm * N + gn] = v;
      }
    }
}

// =====================  flash attention  =====================
// One wave per (head, 16-query tile). Key blocks of 32 (=WMMA K).
// Q,K,V,O: [S, 768] row-major (head h occupies cols h*64..h*64+63).
__global__ __launch_bounds__(256) void k_attn(const float* __restrict__ Q,
                                              const float* __restrict__ K,
                                              const float* __restrict__ V,
                                              float* __restrict__ O) {
  __shared__ __align__(16) _Float16 plds[8][16][32];  // per-wave P staging (64B rows)
  int tid = threadIdx.x, lane = tid & 31, wv = tid >> 5;
  int lh = lane >> 4, lm = lane & 15;
  int gw = blockIdx.x * 8 + wv;          // 0 .. H*(S/16)-1
  int h  = gw >> 7;                      // /128
  int qt = gw & 127;
  const float scale = 0.125f;            // 1/sqrt(64)
  const size_t hcol = (size_t)h * HDIM;

  // Q fragments via float4 (global_load_b128) + cvt
  int qrow = qt * 16 + lm;
  const float* qp = Q + (size_t)qrow * DM + hcol;
  v16h qf0 = concat8(cvt8(*(const v4f*)(qp + 8 * lh),      *(const v4f*)(qp + 8 * lh + 4)),
                     cvt8(*(const v4f*)(qp + 16 + 8 * lh), *(const v4f*)(qp + 16 + 8 * lh + 4)));
  v16h qf1 = concat8(cvt8(*(const v4f*)(qp + 32 + 8 * lh),      *(const v4f*)(qp + 32 + 8 * lh + 4)),
                     cvt8(*(const v4f*)(qp + 48 + 8 * lh), *(const v4f*)(qp + 48 + 8 * lh + 4)));

  v8f acc[4]; acc[0] = v8f{}; acc[1] = v8f{}; acc[2] = v8f{}; acc[3] = v8f{};
  float mrow[8], lrow[8];
#pragma unroll
  for (int r = 0; r < 8; ++r) { mrow[r] = -INFINITY; lrow[r] = 0.f; }

  int jbmax = (qt * 16) / 32;            // inclusive; covers diagonal
  for (int jb = 0; jb <= jbmax; ++jb) {
    int kbase = jb * 32;
    v8f s[2];
#pragma unroll
    for (int n = 0; n < 2; ++n) {        // scores: Q(16x64) @ K^T(64x16)
      int kcol = kbase + n * 16 + lm;    // key position = B-matrix N
      const float* kp = K + (size_t)kcol * DM + hcol + 8 * lh;  // B-map: K=8*half+e (e<8), 16+8*half+(e-8)
      v16h kf0 = concat8(cvt8(*(const v4f*)(kp),     *(const v4f*)(kp + 4)),
                         cvt8(*(const v4f*)(kp + 16), *(const v4f*)(kp + 20)));
      v16h kf1 = concat8(cvt8(*(const v4f*)(kp + 32), *(const v4f*)(kp + 36)),
                         cvt8(*(const v4f*)(kp + 48), *(const v4f*)(kp + 52)));
      v8f c = {};
      c = __builtin_amdgcn_wmma_f32_16x16x32_f16(false, qf0, false, kf0, (short)0, c, false, false);
      c = __builtin_amdgcn_wmma_f32_16x16x32_f16(false, qf1, false, kf1, (short)0, c, false, false);
      s[n] = c;
    }
    // scale + causal mask in C-layout coordinates
#pragma unroll
    for (int n = 0; n < 2; ++n)
#pragma unroll
      for (int r = 0; r < 8; ++r) {
        int qi = qt * 16 + r + 8 * lh;
        int ki = kbase + n * 16 + lm;
        float v = s[n][r] * scale;
        s[n][r] = (ki <= qi) ? v : -INFINITY;
      }
    // online softmax (row = 16 lanes of one half; xor masks <16 stay in-half)
#pragma unroll
    for (int r = 0; r < 8; ++r) {
      float m = fmaxf(s[0][r], s[1][r]);
#pragma unroll
      for (int off = 1; off < 16; off <<= 1) m = fmaxf(m, __shfl_xor(m, off, 32));
      float newm = fmaxf(mrow[r], m);
      float corr = __expf(mrow[r] - newm);
      acc[0][r] *= corr; acc[1][r] *= corr; acc[2][r] *= corr; acc[3][r] *= corr;
      float p0 = __expf(s[0][r] - newm);
      float p1 = __expf(s[1][r] - newm);
      s[0][r] = p0; s[1][r] = p1;
      float rs = p0 + p1;
#pragma unroll
      for (int off = 1; off < 16; off <<= 1) rs += __shfl_xor(rs, off, 32);
      lrow[r] = lrow[r] * corr + rs;
      mrow[r] = newm;
    }
    // P: C-layout -> A-layout via per-wave LDS (in-order DS within wave; no barrier needed)
#pragma unroll
    for (int n = 0; n < 2; ++n)
#pragma unroll
      for (int r = 0; r < 8; ++r)
        plds[wv][r + 8 * lh][n * 16 + lm] = (_Float16)s[n][r];
    v16h pf = concat8(*(const v8h*)&plds[wv][lm][16 * lh],
                      *(const v8h*)&plds[wv][lm][16 * lh + 8]);   // two ds_load_b128
    // acc += P(16x32) @ V(32x64)
#pragma unroll
    for (int n = 0; n < 4; ++n) {
      v16h vf;
#pragma unroll
      for (int e = 0; e < 16; ++e) {
        int kk = kbase + e + 16 * lh;    // key position = B-matrix K (stride-DM gather)
        vf[e] = (_Float16)V[(size_t)kk * DM + hcol + n * 16 + lm];
      }
      acc[n] = __builtin_amdgcn_wmma_f32_16x16x32_f16(false, pf, false, vf, (short)0, acc[n], false, false);
    }
  }
  // normalize + store
#pragma unroll
  for (int n = 0; n < 4; ++n)
#pragma unroll
    for (int r = 0; r < 8; ++r) {
      int qi = qt * 16 + r + 8 * lh;
      O[(size_t)qi * DM + hcol + n * 16 + lm] = acc[n][r] / lrow[r];
    }
}

// =====================  orchestration  =====================
extern "C" void kernel_launch(void* const* d_in, const int* in_sizes, int n_in,
                              void* d_out, int out_size, void* d_ws, size_t ws_size,
                              hipStream_t stream) {
  const int*   ids   = (const int*)  d_in[0];
  const float* wte   = (const float*)d_in[1];
  const float* wpe   = (const float*)d_in[2];
  const float* ln1_g = (const float*)d_in[3];
  const float* ln1_b = (const float*)d_in[4];
  const float* q_w   = (const float*)d_in[5];
  const float* q_b   = (const float*)d_in[6];
  const float* k_w   = (const float*)d_in[7];
  const float* k_b   = (const float*)d_in[8];
  const float* v_w   = (const float*)d_in[9];
  const float* v_b   = (const float*)d_in[10];
  const float* o_w   = (const float*)d_in[11];
  const float* o_b   = (const float*)d_in[12];
  const float* ln2_g = (const float*)d_in[13];
  const float* ln2_b = (const float*)d_in[14];
  const float* fc1_w = (const float*)d_in[15];
  const float* fc1_b = (const float*)d_in[16];
  const float* fc2_w = (const float*)d_in[17];
  const float* fc2_b = (const float*)d_in[18];
  const float* lnf_g = (const float*)d_in[19];
  const float* lnf_b = (const float*)d_in[20];
  const float* lm_w  = (const float*)d_in[21];
  float* out = (float*)d_out;

  float* ws = (float*)d_ws;
  float* x  = ws;              // [S,D] residual stream
  float* hb = ws + (size_t)SD; // [S,D] LN output
  float* qb = ws + 2*(size_t)SD;
  float* kb = ws + 3*(size_t)SD;
  float* vb = ws + 4*(size_t)SD;
  float* ao = ws + 5*(size_t)SD;
  float* ff = ws + 6*(size_t)SD; // [S,FF]

  dim3 gD(SEQ / BM, DM  / BN);   // 16 x 6
  dim3 gF(SEQ / BM, FFD / BN);   // 16 x 24
  dim3 gV(SEQ / BM, VOC / BN);   // 16 x 393

  k_embed<<<SEQ, 256, 0, stream>>>(ids, wte, wpe, x);

  for (int l = 0; l < 6; ++l) {
    size_t wo  = (size_t)l * DM * DM;
    size_t bo  = (size_t)l * DM;
    size_t w1o = (size_t)l * DM * FFD;
    size_t b1o = (size_t)l * FFD;
    size_t w2o = (size_t)l * FFD * DM;

    k_layernorm<<<SEQ, 256, 0, stream>>>(x, ln1_g + bo, ln1_b + bo, hb);
    k_gemm<<<gD, 256, 0, stream>>>(hb, q_w + wo, q_b + bo, nullptr, qb, SEQ, DM, DM, 0);
    k_gemm<<<gD, 256, 0, stream>>>(hb, k_w + wo, k_b + bo, nullptr, kb, SEQ, DM, DM, 0);
    k_gemm<<<gD, 256, 0, stream>>>(hb, v_w + wo, v_b + bo, nullptr, vb, SEQ, DM, DM, 0);
    k_attn<<<NH * (SEQ / 16) / 8, 256, 0, stream>>>(qb, kb, vb, ao);
    k_gemm<<<gD, 256, 0, stream>>>(ao, o_w + wo, o_b + bo, x, x, SEQ, DM, DM, 0);
    k_layernorm<<<SEQ, 256, 0, stream>>>(x, ln2_g + bo, ln2_b + bo, hb);
    k_gemm<<<gF, 256, 0, stream>>>(hb, fc1_w + w1o, fc1_b + b1o, nullptr, ff, SEQ, FFD, DM, GELU_FLAG);
    k_gemm<<<gD, 256, 0, stream>>>(ff, fc2_w + w2o, fc2_b + bo, x, x, SEQ, DM, FFD, 0);
  }

  k_layernorm<<<SEQ, 256, 0, stream>>>(x, lnf_g, lnf_b, hb);
  k_gemm<<<gV, 256, 0, stream>>>(hb, lm_w, nullptr, nullptr, out, SEQ, VOC, DM, 0);
}
